// Gated_attention_cross_interaction_88682484727827
// MI455X (gfx1250) — matrix operand order repaired and checked by reading, not compiled, not run on hardware
//
#include <hip/hip_runtime.h>

typedef _Float16 f16;
typedef __attribute__((ext_vector_type(16))) _Float16 v16h;
typedef __attribute__((ext_vector_type(8)))  _Float16 v8h;
typedef __attribute__((ext_vector_type(8)))  float    v8f;
typedef __attribute__((ext_vector_type(4)))  float    v4f;

#define HDIM 1024
#define LQ   512
#define LDD  4096
#define NB   8
#define DS1  (HDIM + 8)   // Dh LDS row stride (f16) -> rows 4 banks apart, conflict-free b128
#define PS   (LQ + 8)     // P  LDS row stride (f16)

union V16 { v16h v; v8h h[2]; };

// A fragment (16x32 f16, M x K), row-major source [M][K]:
// lanes 0-15: row=lane, K = {k..k+7, k+16..k+23}; lanes 16-31: row=lane-16, K = {k+8..k+15, k+24..k+31}
__device__ __forceinline__ v16h ldA(const f16* base, int stride, int k, int lane) {
  int row = lane & 15;
  int kk  = k + ((lane >> 4) << 3);
  const f16* p = base + (size_t)row * stride + kk;
  V16 u;
  u.h[0] = *(const v8h*)(p);
  u.h[1] = *(const v8h*)(p + 16);
  return u.v;
}

// B fragment (32x16 f16, K x N) loaded from row-major B^T storage [N][K]:
// lanes 0-15: col=nbase+lane, K = {k..k+7, k+16..k+23} ; lanes 16-31: col=nbase+lane-16, K = {k+8..k+15, k+24..k+31}
__device__ __forceinline__ v16h ldB(const f16* base, int stride, int nbase, int k, int lane) {
  int col = nbase + (lane & 15);
  int kk  = k + ((lane >> 4) << 3);
  const f16* p = base + (size_t)col * stride + kk;
  V16 u;
  u.h[0] = *(const v8h*)(p);
  u.h[1] = *(const v8h*)(p + 16);
  return u.v;
}

__device__ __forceinline__ v8f wmma_f16(v16h a, v16h b, v8f c) {
  return __builtin_amdgcn_wmma_f32_16x16x32_f16(false, a, false, b, (short)0, c, false, false);
}

// ---------------- pre-kernel 1: Qh = l2norm(Q) as f16, [B*Lq][H] ----------------
__global__ __launch_bounds__(256) void k_normq(const float* __restrict__ Q, f16* __restrict__ Qh) {
  int lane = threadIdx.x & 31;
  int wv   = threadIdx.x >> 5;
  int row  = blockIdx.x * 8 + wv;                 // < NB*LQ = 4096
  const float* q = Q + (size_t)row * HDIM + lane * 32;
  float v[32];
#pragma unroll
  for (int j = 0; j < 8; ++j) {
    v4f t = __builtin_nontemporal_load((const v4f*)(q + 4 * j));
    v[4*j+0] = t[0]; v[4*j+1] = t[1]; v[4*j+2] = t[2]; v[4*j+3] = t[3];
  }
  float ss = 0.f;
#pragma unroll
  for (int i = 0; i < 32; ++i) ss += v[i] * v[i];
#pragma unroll
  for (int m = 1; m <= 16; m <<= 1) ss += __shfl_xor(ss, m, 32);
  float rn = 1.0f / fmaxf(sqrtf(ss), 1e-8f);
  f16* o = Qh + (size_t)row * HDIM + lane * 32;
#pragma unroll
  for (int j = 0; j < 4; ++j) {
    v8h h;
#pragma unroll
    for (int e = 0; e < 8; ++e) h[e] = (f16)(v[8*j+e] * rn);
    *(v8h*)(o + 8 * j) = h;   // Qh is hot data: keep default (L2-resident) policy
  }
}

// ---------------- pre-kernel 2: Qt = Q^T as f16, [B][H][Lq] ----------------
__global__ __launch_bounds__(256) void k_transq(const float* __restrict__ Q, f16* __restrict__ Qt) {
  __shared__ f16 t[32][33];
  int tx = threadIdx.x & 31, ty = threadIdx.x >> 5;   // ty 0..7
  int h0 = blockIdx.x * 32, lq0 = blockIdx.y * 32, b = blockIdx.z;
  const float* q = Q + (size_t)b * LQ * HDIM;
#pragma unroll
  for (int j = 0; j < 32; j += 8)
    t[ty + j][tx] = (f16)__builtin_nontemporal_load(&q[(size_t)(lq0 + ty + j) * HDIM + h0 + tx]);
  __syncthreads();
  f16* o = Qt + (size_t)b * HDIM * LQ;
#pragma unroll
  for (int j = 0; j < 32; j += 8)
    o[(size_t)(h0 + ty + j) * LQ + lq0 + tx] = t[tx][ty + j];  // Qt hot: default policy
}

// ---------------- main kernel: per 32 D-rows, flash-attention style ----------------
__global__ __launch_bounds__(256) void k_attn(const float* __restrict__ D,
                                              const f16* __restrict__ Qh,
                                              const f16* __restrict__ Qt,
                                              float* __restrict__ out) {
  __shared__ f16   sDh[32 * DS1];     // normalized D tile (32 rows), f16
  __shared__ f16   sP [32 * PS];      // exp(S - max), f16
  __shared__ float sRed[8 * 32];      // per-wave row max, then row sum
  __shared__ float sG[32];            // global row max, then 1/rowsum

  int tid  = threadIdx.x;
  int lane = tid & 31;
  int wv   = tid >> 5;
  int bd   = blockIdx.x;              // 0 .. 1023
  int b    = bd >> 7;                 // / (LDD/32)
  int d0   = (bd & 127) << 5;

  const float* Db  = D  + (size_t)b * LDD * HDIM;
  const f16*   Qhb = Qh + (size_t)b * LQ  * HDIM;
  const f16*   Qtb = Qt + (size_t)b * HDIM * LQ;

  // ---- Phase A: normalize 32 D rows into LDS as f16 (4 rows per wave) ----
  // D is touched once per tile here + once in the epilogue (tile stays in L2
  // between the two; whole-array D must NOT evict Qh/Qt -> stream it NT).
#pragma unroll
  for (int rr = 0; rr < 4; ++rr) {
    int row = wv * 4 + rr;
    const float* dr = Db + (size_t)(d0 + row) * HDIM + lane * 32;
    float v[32];
#pragma unroll
    for (int j = 0; j < 8; ++j) {
      v4f t = __builtin_nontemporal_load((const v4f*)(dr + 4 * j));
      v[4*j+0] = t[0]; v[4*j+1] = t[1]; v[4*j+2] = t[2]; v[4*j+3] = t[3];
    }
    float ss = 0.f;
#pragma unroll
    for (int i = 0; i < 32; ++i) ss += v[i] * v[i];
#pragma unroll
    for (int m = 1; m <= 16; m <<= 1) ss += __shfl_xor(ss, m, 32);
    float rn = 1.0f / fmaxf(sqrtf(ss), 1e-8f);
    f16* dst = &sDh[row * DS1 + lane * 32];
#pragma unroll
    for (int j = 0; j < 4; ++j) {
      v8h h;
#pragma unroll
      for (int e = 0; e < 8; ++e) h[e] = (f16)(v[8*j+e] * rn);
      *(v8h*)(dst + 8 * j) = h;
    }
  }
  __syncthreads();

  // ---- Phase B: S = Dh * Qh^T ; wave: 2 M-tiles x 4 N-tiles (64 Lq cols), full K ----
  int q0 = wv * 64;
  v8f S0[4], S1[4];
#pragma unroll
  for (int t = 0; t < 4; ++t)
#pragma unroll
    for (int r = 0; r < 8; ++r) { S0[t][r] = 0.f; S1[t][r] = 0.f; }

  for (int kk = 0; kk < HDIM; kk += 32) {
    v16h a0 = ldA(sDh,            DS1, kk, lane);
    v16h a1 = ldA(sDh + 16 * DS1, DS1, kk, lane);
#pragma unroll
    for (int t = 0; t < 4; ++t) {
      v16h bf = ldB(Qhb, HDIM, q0 + t * 16, kk, lane);
      S0[t] = wmma_f16(a0, bf, S0[t]);
      S1[t] = wmma_f16(a1, bf, S1[t]);
    }
  }

  // ---- softmax: exact two-pass over the 32-row tile ----
  int rowOff = (lane >> 4) * 8;       // accum elem r -> tile row m*16 + rowOff + r
  float mrow[2][8];
#pragma unroll
  for (int r = 0; r < 8; ++r) {
    float m0 = fmaxf(fmaxf(S0[0][r], S0[1][r]), fmaxf(S0[2][r], S0[3][r]));
    float m1 = fmaxf(fmaxf(S1[0][r], S1[1][r]), fmaxf(S1[2][r], S1[3][r]));
#pragma unroll
    for (int msk = 1; msk <= 8; msk <<= 1) {
      m0 = fmaxf(m0, __shfl_xor(m0, msk, 32));
      m1 = fmaxf(m1, __shfl_xor(m1, msk, 32));
    }
    mrow[0][r] = m0; mrow[1][r] = m1;
  }
  if ((lane & 15) == 0) {
#pragma unroll
    for (int m = 0; m < 2; ++m)
#pragma unroll
      for (int r = 0; r < 8; ++r) sRed[wv * 32 + m * 16 + rowOff + r] = mrow[m][r];
  }
  __syncthreads();
  if (wv == 0) {                       // 32 lanes handle 32 rows
    float g = sRed[lane];
#pragma unroll
    for (int w = 1; w < 8; ++w) g = fmaxf(g, sRed[w * 32 + lane]);
    sG[lane] = g;
  }
  __syncthreads();

  float gm[2][8];
#pragma unroll
  for (int r = 0; r < 8; ++r) {
    gm[0][r] = sG[rowOff + r];
    gm[1][r] = sG[16 + rowOff + r];
  }

  // exp, row sums, store P (unnormalized) to LDS as f16
  float srow[2][8];
#pragma unroll
  for (int r = 0; r < 8; ++r) { srow[0][r] = 0.f; srow[1][r] = 0.f; }
#pragma unroll
  for (int t = 0; t < 4; ++t) {
#pragma unroll
    for (int r = 0; r < 8; ++r) {
      float p0 = __expf(S0[t][r] - gm[0][r]);
      float p1 = __expf(S1[t][r] - gm[1][r]);
      srow[0][r] += p0;
      srow[1][r] += p1;
      int col = q0 + t * 16 + (lane & 15);
      sP[(rowOff + r) * PS + col]        = (f16)p0;
      sP[(16 + rowOff + r) * PS + col]   = (f16)p1;
    }
  }
#pragma unroll
  for (int r = 0; r < 8; ++r) {
#pragma unroll
    for (int msk = 1; msk <= 8; msk <<= 1) {
      srow[0][r] += __shfl_xor(srow[0][r], msk, 32);
      srow[1][r] += __shfl_xor(srow[1][r], msk, 32);
    }
  }
  if ((lane & 15) == 0) {
#pragma unroll
    for (int m = 0; m < 2; ++m)
#pragma unroll
      for (int r = 0; r < 8; ++r) sRed[wv * 32 + m * 16 + rowOff + r] = srow[m][r];
  }
  __syncthreads();
  if (wv == 0) {
    float s = 0.f;
#pragma unroll
    for (int w = 0; w < 8; ++w) s += sRed[w * 32 + lane];
    sG[lane] = 1.0f / s;
  }
  __syncthreads();

  float rl[2][8];
#pragma unroll
  for (int r = 0; r < 8; ++r) {
    rl[0][r] = sG[rowOff + r];
    rl[1][r] = sG[16 + rowOff + r];
  }

  // ---- Phase C: O = P * Q ; wave: 2 M-tiles x 4 N-tiles (64 H cols), 2 passes over H ----
  const float* Dt = Db + (size_t)d0 * HDIM;
  float*       ob = out + ((size_t)b * LDD + d0) * HDIM;

#pragma unroll
  for (int pass = 0; pass < 2; ++pass) {
    int h0 = wv * 64 + pass * 512;
    v8f O0[4], O1[4];
#pragma unroll
    for (int t = 0; t < 4; ++t)
#pragma unroll
      for (int r = 0; r < 8; ++r) { O0[t][r] = 0.f; O1[t][r] = 0.f; }

    for (int kk = 0; kk < LQ; kk += 32) {
      v16h a0 = ldA(sP,           PS, kk, lane);
      v16h a1 = ldA(sP + 16 * PS, PS, kk, lane);
#pragma unroll
      for (int t = 0; t < 4; ++t) {
        v16h bf = ldB(Qtb, LQ, h0 + t * 16, kk, lane);
        O0[t] = wmma_f16(a0, bf, O0[t]);
        O1[t] = wmma_f16(a1, bf, O1[t]);
      }
    }

    // epilogue for this H half: out = O / rowsum + D  (stream NT both ways)
#pragma unroll
    for (int t = 0; t < 4; ++t) {
      int col = h0 + t * 16 + (lane & 15);
#pragma unroll
      for (int r = 0; r < 8; ++r) {
        size_t i0 = (size_t)(rowOff + r) * HDIM + col;
        size_t i1 = (size_t)(16 + rowOff + r) * HDIM + col;
        float d0v = __builtin_nontemporal_load(&Dt[i0]);
        float d1v = __builtin_nontemporal_load(&Dt[i1]);
        __builtin_nontemporal_store(O0[t][r] * rl[0][r] + d0v, &ob[i0]);
        __builtin_nontemporal_store(O1[t][r] * rl[1][r] + d1v, &ob[i1]);
      }
    }
  }
}

extern "C" void kernel_launch(void* const* d_in, const int* in_sizes, int n_in,
                              void* d_out, int out_size, void* d_ws, size_t ws_size,
                              hipStream_t stream) {
  const float* Q = (const float*)d_in[0];   // [8, 512, 1024] f32
  const float* D = (const float*)d_in[1];   // [8, 4096, 1024] f32
  float* out = (float*)d_out;               // [8, 4096, 1024] f32

  f16* Qh = (f16*)d_ws;                        // 8 MB
  f16* Qt = Qh + (size_t)NB * LQ * HDIM;       // 8 MB

  k_normq <<<(NB * LQ) / 8, 256, 0, stream>>>(Q, Qh);
  k_transq<<<dim3(HDIM / 32, LQ / 32, NB), 256, 0, stream>>>(Q, Qt);
  k_attn  <<<NB * (LDD / 32), 256, 0, stream>>>(D, Qh, Qt, out);
}
